// Actor_50517405335740
// MI455X (gfx1250) — matrix operand order repaired and checked by reading, not compiled, not run on hardware
//
#include <hip/hip_runtime.h>
#include <hip/hip_bf16.h>

// ---------------------------------------------------------------------------
// Problem constants (from reference setup_inputs)
// ---------------------------------------------------------------------------
#define BB    4096   // batch
#define OBS   256
#define ACT   32
#define HH    512
#define MM    16
#define HM    8192   // H*M
#define LN_EPS 1e-5f
#define LOG_STD_MAX 2.0f
#define LOG_STD_MIN -5.0f

typedef __attribute__((ext_vector_type(16))) __bf16 v16bf;
typedef __attribute__((ext_vector_type(8)))  float  v8f;
typedef __attribute__((ext_vector_type(4)))  int    v4i;

union FragAB { v16bf v; float4 q[2]; };
union FragC  { v8f   v; float  f[8]; };

// CDNA5 async global->LDS staging (ASYNCcnt path), if the toolchain declares it
#if __has_builtin(__builtin_amdgcn_global_load_async_to_lds_b128) && \
    __has_builtin(__builtin_amdgcn_s_wait_asynccnt)
#define ASYNC_STAGE 1
#define GAS1(p) ((__attribute__((address_space(1))) v4i*)(p))
#define LAS3(p) ((__attribute__((address_space(3))) v4i*)(p))
#else
#define ASYNC_STAGE 0
#endif

// ---------------------------------------------------------------------------
// Elementwise f32 -> bf16
// ---------------------------------------------------------------------------
__global__ __launch_bounds__(256) void convert_bf16_kernel(const float* __restrict__ src,
                                                           __hip_bfloat16* __restrict__ dst,
                                                           size_t n) {
    size_t i = (size_t)blockIdx.x * blockDim.x + threadIdx.x;
    if (i < n) dst[i] = __float2bfloat16(src[i]);
}

// ---------------------------------------------------------------------------
// f32 [K][N] row-major  ->  bf16 [N][K] row-major (transpose + convert)
// ---------------------------------------------------------------------------
__global__ __launch_bounds__(256) void transpose_convert_kernel(const float* __restrict__ src,
                                                                __hip_bfloat16* __restrict__ dst,
                                                                int K, int N) {
    __shared__ float tile[32][33];
    int nb = blockIdx.x * 32;
    int kb = blockIdx.y * 32;
    for (int i = threadIdx.y; i < 32; i += 8)
        tile[i][threadIdx.x] = src[(size_t)(kb + i) * N + nb + threadIdx.x];
    __syncthreads();
    for (int i = threadIdx.y; i < 32; i += 8)
        dst[(size_t)(nb + i) * K + kb + threadIdx.x] = __float2bfloat16(tile[threadIdx.x][i]);
}

// ---------------------------------------------------------------------------
// Gating: softmax -> top-k -> renormalize -> scatter. One wave32 per row.
// ---------------------------------------------------------------------------
__global__ __launch_bounds__(32) void gate_kernel(const float* __restrict__ x,
                                                  const float* __restrict__ gW,
                                                  const float* __restrict__ gb,
                                                  const int*  __restrict__ topk,
                                                  float* __restrict__ scores) {
    int b    = blockIdx.x;
    int lane = threadIdx.x;
    int e    = lane & 15;
    int half = lane >> 4;
    const float* xr = x + (size_t)b * OBS;

    float s = 0.f;
    for (int k = half * 128; k < half * 128 + 128; ++k)
        s += xr[k] * gW[k * MM + e];
    s += __shfl_xor(s, 16);          // both halves now hold full dot for expert e
    float logit = s + gb[e];

    // softmax over 16 experts (mirrored in lanes 0-15 / 16-31)
    float mx = logit;
    for (int m = 1; m < 16; m <<= 1) mx = fmaxf(mx, __shfl_xor(mx, m));
    float p = expf(logit - mx);
    float sum = p;
    for (int m = 1; m < 16; m <<= 1) sum += __shfl_xor(sum, m);
    float prob = p / sum;

    // iterative top-k argmax
    int   K    = topk[0];
    float work = prob;
    float ssum = 0.f;
    int   sel  = 0;
    for (int it = 0; it < K; ++it) {
        float v = work; int idx = e;
        for (int m = 1; m < 16; m <<= 1) {
            float v2 = __shfl_xor(v, m);
            int   i2 = __shfl_xor(idx, m);
            if (v2 > v || (v2 == v && i2 < idx)) { v = v2; idx = i2; }
        }
        ssum += v;
        if (e == idx) { sel = 1; work = -1.f; }
    }
    if (lane < 16) scores[(size_t)b * MM + e] = sel ? (prob / ssum) : 0.f;
}

// ---------------------------------------------------------------------------
// GEMM: C[M,N] = A[M,K] @ B[K,N] + bias[N], A bf16 row-major, B given as
// Bt bf16 [N][K]. WMMA f32_16x16x32_bf16, 128x128x32 block tile, 8 waves.
// Double-buffered LDS with GLOBAL_LOAD_ASYNC_TO_LDS_B128 when available.
// ---------------------------------------------------------------------------
__global__ __launch_bounds__(256) void gemm_bf16_kernel(const __hip_bfloat16* __restrict__ A,
                                                        const __hip_bfloat16* __restrict__ Bt,
                                                        const float* __restrict__ bias,
                                                        float* __restrict__ C,
                                                        int Kdim, int N) {
#if ASYNC_STAGE
    __shared__ __align__(16) __hip_bfloat16 As[2][128][40];
    __shared__ __align__(16) __hip_bfloat16 Bs[2][128][40];
#else
    __shared__ __align__(16) __hip_bfloat16 As[1][128][40];
    __shared__ __align__(16) __hip_bfloat16 Bs[1][128][40];
#endif

    const int tid  = threadIdx.x;
    const int wid  = tid >> 5;
    const int lane = tid & 31;
    const int wm   = wid >> 2;           // 0..1  (64-row slice)
    const int wn   = wid & 3;            // 0..3  (32-col slice)
    const int lrow = lane & 15;
    const int hi   = lane >> 4;          // 0 or 1

    const int rowBase = blockIdx.y * 128;
    const int colBase = blockIdx.x * 128;

    FragC acc[4][2];
#pragma unroll
    for (int i = 0; i < 4; ++i)
#pragma unroll
        for (int j = 0; j < 2; ++j)
            acc[i][j].v = (v8f){0.f, 0.f, 0.f, 0.f, 0.f, 0.f, 0.f, 0.f};

    const int sr = tid >> 1;             // staging row 0..127
    const int sh = (tid & 1) * 16;       // staging k-offset 0 or 16
    const __hip_bfloat16* gA = A  + (size_t)(rowBase + sr) * Kdim + sh;
    const __hip_bfloat16* gB = Bt + (size_t)(colBase + sr) * Kdim + sh;

#if ASYNC_STAGE
    // -------- async double-buffered pipeline --------
    // prologue: tile 0 -> buffer 0 (4 x b128 per thread, tracked by ASYNCcnt)
    __builtin_amdgcn_global_load_async_to_lds_b128(GAS1(gA),     LAS3(&As[0][sr][sh]),     0, 0);
    __builtin_amdgcn_global_load_async_to_lds_b128(GAS1(gA + 8), LAS3(&As[0][sr][sh + 8]), 0, 0);
    __builtin_amdgcn_global_load_async_to_lds_b128(GAS1(gB),     LAS3(&Bs[0][sr][sh]),     0, 0);
    __builtin_amdgcn_global_load_async_to_lds_b128(GAS1(gB + 8), LAS3(&Bs[0][sr][sh + 8]), 0, 0);
    int buf = 0;
    for (int k0 = 0; k0 < Kdim; k0 += 32) {
        const bool more = (k0 + 32) < Kdim;
        if (more) {
            const int nb = buf ^ 1;
            const int kn = k0 + 32;
            __builtin_amdgcn_global_load_async_to_lds_b128(GAS1(gA + kn),     LAS3(&As[nb][sr][sh]),     0, 0);
            __builtin_amdgcn_global_load_async_to_lds_b128(GAS1(gA + kn + 8), LAS3(&As[nb][sr][sh + 8]), 0, 0);
            __builtin_amdgcn_global_load_async_to_lds_b128(GAS1(gB + kn),     LAS3(&Bs[nb][sr][sh]),     0, 0);
            __builtin_amdgcn_global_load_async_to_lds_b128(GAS1(gB + kn + 8), LAS3(&Bs[nb][sr][sh + 8]), 0, 0);
            __builtin_amdgcn_s_wait_asynccnt(4);   // current tile's 4 ops done
        } else {
            __builtin_amdgcn_s_wait_asynccnt(0);
        }
        __syncthreads();                           // all waves' tile data visible

        FragAB afrag[4];
        const int akb = hi * 8;
#pragma unroll
        for (int i = 0; i < 4; ++i) {
            int r = wm * 64 + i * 16 + lrow;
            afrag[i].q[0] = *(const float4*)&As[buf][r][akb];
            afrag[i].q[1] = *(const float4*)&As[buf][r][akb + 16];
        }
#pragma unroll
        for (int j = 0; j < 2; ++j) {
            FragAB bfrag;
            int n  = wn * 32 + j * 16 + lrow;
            int kb = hi * 8;
            bfrag.q[0] = *(const float4*)&Bs[buf][n][kb];
            bfrag.q[1] = *(const float4*)&Bs[buf][n][kb + 16];
#pragma unroll
            for (int i = 0; i < 4; ++i) {
                acc[i][j].v = __builtin_amdgcn_wmma_f32_16x16x32_bf16(
                    false, afrag[i].v, false, bfrag.v,
                    (short)0, acc[i][j].v, false, false);
            }
        }
        __syncthreads();                           // done reading before overwrite
        buf ^= 1;
    }
#else
    // -------- synchronous fallback (load -> ds_store) --------
    for (int k0 = 0; k0 < Kdim; k0 += 32) {
        {
            const uint4* sa = (const uint4*)(gA + k0);
            uint4 a0 = sa[0], a1 = sa[1];
            *(uint4*)&As[0][sr][sh]     = a0;
            *(uint4*)&As[0][sr][sh + 8] = a1;
            const uint4* sb = (const uint4*)(gB + k0);
            uint4 b0 = sb[0], b1 = sb[1];
            *(uint4*)&Bs[0][sr][sh]     = b0;
            *(uint4*)&Bs[0][sr][sh + 8] = b1;
        }
        if (k0 + 32 < Kdim) {            // hint next tile into GL2
            __builtin_prefetch(gA + k0 + 32, 0, 1);
            __builtin_prefetch(gB + k0 + 32, 0, 1);
        }
        __syncthreads();

        FragAB afrag[4];
        const int akb = hi * 8;
#pragma unroll
        for (int i = 0; i < 4; ++i) {
            int r = wm * 64 + i * 16 + lrow;
            afrag[i].q[0] = *(const float4*)&As[0][r][akb];
            afrag[i].q[1] = *(const float4*)&As[0][r][akb + 16];
        }
#pragma unroll
        for (int j = 0; j < 2; ++j) {
            FragAB bfrag;
            int n  = wn * 32 + j * 16 + lrow;
            int kb = hi * 8;
            bfrag.q[0] = *(const float4*)&Bs[0][n][kb];
            bfrag.q[1] = *(const float4*)&Bs[0][n][kb + 16];
#pragma unroll
            for (int i = 0; i < 4; ++i) {
                acc[i][j].v = __builtin_amdgcn_wmma_f32_16x16x32_bf16(
                    false, afrag[i].v, false, bfrag.v,
                    (short)0, acc[i][j].v, false, false);
            }
        }
        __syncthreads();
    }
#endif

    // epilogue: D(m,n): lane -> n = lrow (+16*hi picks m+8), vgpr r -> m = r + 8*hi
#pragma unroll
    for (int j = 0; j < 2; ++j) {
        int n  = colBase + wn * 32 + j * 16 + lrow;
        float bn = bias[n];
#pragma unroll
        for (int i = 0; i < 4; ++i) {
            int mb = rowBase + wm * 64 + i * 16 + hi * 8;
#pragma unroll
            for (int r = 0; r < 8; ++r)
                C[(size_t)(mb + r) * N + n] = acc[i][j].f[r] + bn;
        }
    }
}

// ---------------------------------------------------------------------------
// Row LayerNorm + ReLU -> bf16. One 256-thread block per row of 8192.
// ---------------------------------------------------------------------------
__global__ __launch_bounds__(256) void ln_relu_kernel(const float* __restrict__ pre,
                                                      const float* __restrict__ scale,
                                                      const float* __restrict__ bias,
                                                      __hip_bfloat16* __restrict__ out) {
    int row = blockIdx.x, tid = threadIdx.x;
    const float* pr = pre + (size_t)row * HM;
    float v[32];
    const float4* p4 = (const float4*)(pr + tid * 32);
#pragma unroll
    for (int i = 0; i < 8; ++i) {
        float4 q = p4[i];
        v[4 * i] = q.x; v[4 * i + 1] = q.y; v[4 * i + 2] = q.z; v[4 * i + 3] = q.w;
    }
    float s = 0.f, ss = 0.f;
#pragma unroll
    for (int i = 0; i < 32; ++i) { s += v[i]; ss += v[i] * v[i]; }
    for (int m = 1; m < 32; m <<= 1) { s += __shfl_xor(s, m); ss += __shfl_xor(ss, m); }
    __shared__ float sums[8], sqs[8];
    if ((tid & 31) == 0) { sums[tid >> 5] = s; sqs[tid >> 5] = ss; }
    __syncthreads();
    float ts = 0.f, tss = 0.f;
#pragma unroll
    for (int w = 0; w < 8; ++w) { ts += sums[w]; tss += sqs[w]; }
    float mu   = ts * (1.f / HM);
    float var  = tss * (1.f / HM) - mu * mu;
    float rstd = rsqrtf(var + LN_EPS);
#pragma unroll
    for (int i = 0; i < 32; ++i) {
        int c = tid * 32 + i;
        float y = (v[i] - mu) * rstd * scale[c] + bias[c];
        out[(size_t)row * HM + c] = __float2bfloat16(fmaxf(y, 0.f));
    }
}

// ---------------------------------------------------------------------------
// Row LayerNorm + ReLU fused with expert-weighted mean:
// mixed[b,h] = (1/M) * sum_m relu(LN(pre))[b, h*M+m] * scores[b,m]
// ---------------------------------------------------------------------------
__global__ __launch_bounds__(256) void ln_relu_mix_kernel(const float* __restrict__ pre,
                                                          const float* __restrict__ scale,
                                                          const float* __restrict__ bias,
                                                          const float* __restrict__ scores,
                                                          float* __restrict__ mixed) {
    int row = blockIdx.x, tid = threadIdx.x;
    __shared__ float sc[MM];
    if (tid < MM) sc[tid] = scores[(size_t)row * MM + tid];
    const float* pr = pre + (size_t)row * HM;
    float v[32];
    const float4* p4 = (const float4*)(pr + tid * 32);
#pragma unroll
    for (int i = 0; i < 8; ++i) {
        float4 q = p4[i];
        v[4 * i] = q.x; v[4 * i + 1] = q.y; v[4 * i + 2] = q.z; v[4 * i + 3] = q.w;
    }
    float s = 0.f, ss = 0.f;
#pragma unroll
    for (int i = 0; i < 32; ++i) { s += v[i]; ss += v[i] * v[i]; }
    for (int m = 1; m < 32; m <<= 1) { s += __shfl_xor(s, m); ss += __shfl_xor(ss, m); }
    __shared__ float sums[8], sqs[8];
    if ((tid & 31) == 0) { sums[tid >> 5] = s; sqs[tid >> 5] = ss; }
    __syncthreads();
    float ts = 0.f, tss = 0.f;
#pragma unroll
    for (int w = 0; w < 8; ++w) { ts += sums[w]; tss += sqs[w]; }
    float mu   = ts * (1.f / HM);
    float var  = tss * (1.f / HM) - mu * mu;
    float rstd = rsqrtf(var + LN_EPS);

    float accA = 0.f, accB = 0.f;
#pragma unroll
    for (int i = 0; i < 16; ++i) {
        int c = tid * 32 + i;
        float y = fmaxf((v[i] - mu) * rstd * scale[c] + bias[c], 0.f);
        accA += y * sc[i];
    }
#pragma unroll
    for (int i = 16; i < 32; ++i) {
        int c = tid * 32 + i;
        float y = fmaxf((v[i] - mu) * rstd * scale[c] + bias[c], 0.f);
        accB += y * sc[i - 16];
    }
    mixed[(size_t)row * HH + tid * 2]     = accA * (1.f / MM);
    mixed[(size_t)row * HH + tid * 2 + 1] = accB * (1.f / MM);
}

// ---------------------------------------------------------------------------
// Heads: mean = mixed@mean_W+b ; log_std = affine(tanh(mixed@ls_W+b))
// One wave per batch row; lane = action index.
// ---------------------------------------------------------------------------
__global__ __launch_bounds__(256) void head_kernel(const float* __restrict__ mixed,
                                                   const float* __restrict__ mW,
                                                   const float* __restrict__ mb,
                                                   const float* __restrict__ lW,
                                                   const float* __restrict__ lb,
                                                   float* __restrict__ out) {
    int g = blockIdx.x * blockDim.x + threadIdx.x;
    int a = g & (ACT - 1);
    int b = g >> 5;                       // ACT == 32 == wave width
    const float* mr = mixed + (size_t)b * HH;
    float am = 0.f, as = 0.f;
    for (int k = 0; k < HH; ++k) {
        float m = mr[k];
        am += m * mW[k * ACT + a];
        as += m * lW[k * ACT + a];
    }
    float mean = am + mb[a];
    float t = tanhf(as + lb[a]);
    float ls = LOG_STD_MIN + 0.5f * (LOG_STD_MAX - LOG_STD_MIN) * (t + 1.f);
    out[(size_t)b * ACT + a]                     = mean;
    out[(size_t)BB * ACT + (size_t)b * ACT + a]  = ls;
}

// ---------------------------------------------------------------------------
// Launch
// ---------------------------------------------------------------------------
extern "C" void kernel_launch(void* const* d_in, const int* in_sizes, int n_in,
                              void* d_out, int out_size, void* d_ws, size_t ws_size,
                              hipStream_t stream) {
    (void)in_sizes; (void)n_in; (void)out_size; (void)ws_size;
    const float* x      = (const float*)d_in[0];
    const float* gW     = (const float*)d_in[1];
    const float* gb     = (const float*)d_in[2];
    const float* fc1W   = (const float*)d_in[3];
    const float* fc1b   = (const float*)d_in[4];
    const float* n1s    = (const float*)d_in[5];
    const float* n1b    = (const float*)d_in[6];
    const float* fc2W   = (const float*)d_in[7];
    const float* fc2b   = (const float*)d_in[8];
    const float* n2s    = (const float*)d_in[9];
    const float* n2b    = (const float*)d_in[10];
    const float* meanW  = (const float*)d_in[11];
    const float* meanb  = (const float*)d_in[12];
    const float* lsW    = (const float*)d_in[13];
    const float* lsb    = (const float*)d_in[14];
    const int*   topk   = (const int*)d_in[15];
    float* out = (float*)d_out;

    // workspace carve-up (256B aligned)
    char* w = (char*)d_ws;
    size_t off = 0;
    auto carve = [&](size_t bytes) -> void* {
        void* p = w + off;
        off = (off + bytes + 255) & ~(size_t)255;
        return p;
    };
    __hip_bfloat16* xb  = (__hip_bfloat16*)carve((size_t)BB * OBS * 2);
    __hip_bfloat16* w1t = (__hip_bfloat16*)carve((size_t)HM * OBS * 2);
    __hip_bfloat16* w2t = (__hip_bfloat16*)carve((size_t)HM * HM * 2);
    float*          pre = (float*)         carve((size_t)BB * HM * 4);   // reused for both GEMMs
    __hip_bfloat16* h1  = (__hip_bfloat16*)carve((size_t)BB * HM * 2);
    float*          sco = (float*)         carve((size_t)BB * MM * 4);
    float*          mix = (float*)         carve((size_t)BB * HH * 4);

    // 1. precision conversion + weight transposes
    {
        size_t n = (size_t)BB * OBS;
        convert_bf16_kernel<<<(unsigned)((n + 255) / 256), 256, 0, stream>>>(x, xb, n);
    }
    transpose_convert_kernel<<<dim3(HM / 32, OBS / 32), dim3(32, 8), 0, stream>>>(fc1W, w1t, OBS, HM);
    transpose_convert_kernel<<<dim3(HM / 32, HM / 32), dim3(32, 8), 0, stream>>>(fc2W, w2t, HM, HM);

    // 2. gating
    gate_kernel<<<BB, 32, 0, stream>>>(x, gW, gb, topk, sco);

    // 3. fc1 GEMM + LN + ReLU
    gemm_bf16_kernel<<<dim3(HM / 128, BB / 128), 256, 0, stream>>>(xb, w1t, fc1b, pre, OBS, HM);
    ln_relu_kernel<<<BB, 256, 0, stream>>>(pre, n1s, n1b, h1);

    // 4. fc2 GEMM + LN + ReLU + expert mixing
    gemm_bf16_kernel<<<dim3(HM / 128, BB / 128), 256, 0, stream>>>(h1, w2t, fc2b, pre, HM, HM);
    ln_relu_mix_kernel<<<BB, 256, 0, stream>>>(pre, n2s, n2b, sco, mix);

    // 5. heads
    head_kernel<<<(BB * ACT) / 256, 256, 0, stream>>>(mix, meanW, meanb, lsW, lsb, out);
}
